// ScaledDotProductAttention_9096740733080
// MI455X (gfx1250) — matrix operand order repaired and checked by reading, not compiled, not run on hardware
//
#include <hip/hip_runtime.h>
#include <hip/hip_bf16.h>

typedef __attribute__((ext_vector_type(16))) _Float16 v16h;
typedef __attribute__((ext_vector_type(8)))  float    v8f;
typedef int v4i __attribute__((vector_size(16)));   // GCC-style, matches builtin proto

#define Bn 2
#define Hn 16
#define Sn 2048
#define Dn 64
#define QT 64
#define KT 32
#define SKW 68   // padded LDS row stride (floats): 272B, 16B-aligned, rows 4 banks apart

#define AS1 __attribute__((address_space(1)))
#define AS3 __attribute__((address_space(3)))

#if defined(__AMDGCN__) && __has_builtin(__builtin_amdgcn_global_load_async_to_lds_b128)
#define HAVE_ASYNC 1
#else
#define HAVE_ASYNC 0
#endif

#if HAVE_ASYNC
#if __has_builtin(__builtin_amdgcn_s_wait_asynccnt)
#define WAIT_ASYNC(n) __builtin_amdgcn_s_wait_asynccnt(n)
#else
#define WAIT_ASYNC(n) asm volatile("s_wait_asynccnt %0" ::"n"(n) : "memory")
#endif
#else
#define WAIT_ASYNC(n) do {} while (0)
#endif

__device__ __forceinline__ float redmax16(float x) {
#pragma unroll
  for (int o = 1; o < 16; o <<= 1) x = fmaxf(x, __shfl_xor(x, o, 32));
  return x;
}
__device__ __forceinline__ float redsum16(float x) {
#pragma unroll
  for (int o = 1; o < 16; o <<= 1) x += __shfl_xor(x, o, 32);
  return x;
}

// Copy one 32x64-float tile (512 float4) global -> LDS with padded rows.
// Async path: GLOBAL_LOAD_ASYNC_TO_LDS_B128 (16B per lane), tracked by ASYNCcnt.
__device__ __forceinline__ void stage_tile(const float4* __restrict__ gsrc,
                                           void* lds_base, int tid) {
#pragma unroll
  for (int j = 0; j < 4; ++j) {
    const int i   = tid + j * 128;
    const int row = i >> 4;      // 16 float4 per 64-float row
    const int c4  = i & 15;
    char* lp = (char*)lds_base + row * (SKW * 4) + c4 * 16;
#if HAVE_ASYNC
    __builtin_amdgcn_global_load_async_to_lds_b128(
        (AS1 v4i*)(uintptr_t)(gsrc + i), (AS3 v4i*)(uintptr_t)lp, 0, 0);
#else
    *(float4*)lp = gsrc[i];
#endif
  }
}

__global__ void zero_f4_kernel(float4* __restrict__ p, size_t n4) {
  size_t i = (size_t)blockIdx.x * blockDim.x + threadIdx.x;
  const size_t stride = (size_t)gridDim.x * blockDim.x;
  const float4 z = {0.f, 0.f, 0.f, 0.f};
  for (; i < n4; i += stride) p[i] = z;
}

#define WMMA_F16(A, B, C) \
  __builtin_amdgcn_wmma_f32_16x16x32_f16(false, (A), false, (B), (short)0, (C), false, false)

__global__ __launch_bounds__(128) void fa_fwd_kernel(const float* __restrict__ Q,
                                                     const float* __restrict__ K,
                                                     const float* __restrict__ V,
                                                     float* __restrict__ Out,
                                                     float* __restrict__ Attn) {
  __shared__ __align__(16) float sKb[2][KT][SKW];   // double-buffered K tiles
  __shared__ __align__(16) float sVb[2][KT][SKW];   // double-buffered V tiles
  __shared__ _Float16 sP[4][16][34];                // per-wave P bounce, 17-dword rows

  const int tid  = threadIdx.x;
  const int wave = tid >> 5;
  const int lane = tid & 31;
  const int half = lane >> 4;
  const int lm   = lane & 15;

  const int qt = blockIdx.x;
  const int bh = blockIdx.y;

  const size_t base = (size_t)bh * Sn * Dn;
  const int qrow0 = qt * QT + wave * 16;
  const int koff  = half * 8;

  // ---- Q strip as two 16x32 f16 A-fragments, pre-scaled by 1/sqrt(D) ----
  const float scale = 0.125f;
  const float* qrow = Q + base + (size_t)(qrow0 + lm) * Dn;
  v16h qa0, qa1;
#pragma unroll
  for (int j = 0; j < 8; ++j) {
    qa0[j]     = (_Float16)(qrow[koff + j] * scale);
    qa0[8 + j] = (_Float16)(qrow[16 + koff + j] * scale);
    qa1[j]     = (_Float16)(qrow[32 + koff + j] * scale);
    qa1[8 + j] = (_Float16)(qrow[48 + koff + j] * scale);
  }

  float mrow[8], lrow[8];
  v8f o0 = {}, o1 = {}, o2 = {}, o3 = {};
#pragma unroll
  for (int r = 0; r < 8; ++r) { mrow[r] = -1e30f; lrow[r] = 0.f; }

  const int nchunks = 2 * qt + 2;   // causal: keys 0 .. qt*64+63
  const float4* gK4 = (const float4*)(K + base);
  const float4* gV4 = (const float4*)(V + base);

  // =================== pass 1: flash accumulation of O ===================
  // prologue: prefetch chunk 0 into buffer 0
  stage_tile(gK4, &sKb[0][0][0], tid);
  stage_tile(gV4, &sVb[0][0][0], tid);

  for (int c = 0; c < nchunks; ++c) {
    const int b  = c & 1;
    const int k0 = c * KT;
    // prefetch next chunk into the other buffer (its previous readers are done:
    // the trailing barrier of iteration c-1 has already been passed)
    if (c + 1 < nchunks) {
      const int off4 = (k0 + KT) * (Dn / 4);
      stage_tile(gK4 + off4, &sKb[b ^ 1][0][0], tid);
      stage_tile(gV4 + off4, &sVb[b ^ 1][0][0], tid);
      WAIT_ASYNC(8);   // in-order: <=8 outstanding => chunk c fully resident
    } else {
      WAIT_ASYNC(0);
    }
    __syncthreads();

    // S = Q K^T for two 16-key subtiles
    v8f s0 = {}, s1 = {};
    {
      v16h kb0, kb1;
      const float* kr = &sKb[b][lm][0];
#pragma unroll
      for (int e = 0; e < 16; ++e) {
        kb0[e] = (_Float16)kr[koff + e + (e >= 8 ? 8 : 0)];
        kb1[e] = (_Float16)kr[32 + koff + e + (e >= 8 ? 8 : 0)];
      }
      s0 = WMMA_F16(qa0, kb0, s0);
      s0 = WMMA_F16(qa1, kb1, s0);
      kr = &sKb[b][16 + lm][0];
#pragma unroll
      for (int e = 0; e < 16; ++e) {
        kb0[e] = (_Float16)kr[koff + e + (e >= 8 ? 8 : 0)];
        kb1[e] = (_Float16)kr[32 + koff + e + (e >= 8 ? 8 : 0)];
      }
      s1 = WMMA_F16(qa0, kb0, s1);
      s1 = WMMA_F16(qa1, kb1, s1);
    }

    // masked online softmax (single max-update per 32-key block)
#pragma unroll
    for (int r = 0; r < 8; ++r) {
      const int row = qrow0 + half * 8 + r;
      const int kg0 = k0 + lm;
      const int kg1 = k0 + 16 + lm;
      float a = (kg0 <= row) ? s0[r] : -1e30f;
      float bb = (kg1 <= row) ? s1[r] : -1e30f;
      float bm   = fmaxf(redmax16(fmaxf(a, bb)), mrow[r]);
      float corr = __expf(mrow[r] - bm);
      float p0 = __expf(a - bm);
      float p1 = __expf(bb - bm);
      float rs = redsum16(p0 + p1);
      lrow[r] = lrow[r] * corr + rs;
      mrow[r] = bm;
      o0[r] *= corr; o1[r] *= corr; o2[r] *= corr; o3[r] *= corr;
      const int rl = half * 8 + r;
      sP[wave][rl][lm]      = (_Float16)p0;
      sP[wave][rl][16 + lm] = (_Float16)p1;
    }

    // rebuild P as 16x32 A-fragment (DS ops are in-order within a wave)
    v16h pa;
    {
      const _Float16* pr = &sP[wave][lm][0];
#pragma unroll
      for (int j = 0; j < 8; ++j) {
        pa[j]     = pr[koff + j];
        pa[8 + j] = pr[16 + koff + j];
      }
    }

    // O += P @ V  (4 d-tiles)
#define PV_TILE(DT, OC)                                                \
    {                                                                  \
      v16h vb;                                                         \
      _Pragma("unroll")                                                \
      for (int e = 0; e < 16; ++e)                                     \
        vb[e] = (_Float16)sVb[b][koff + e + (e >= 8 ? 8 : 0)][(DT) * 16 + lm]; \
      OC = WMMA_F16(pa, vb, OC);                                       \
    }
    PV_TILE(0, o0)
    PV_TILE(1, o1)
    PV_TILE(2, o2)
    PV_TILE(3, o3)
#undef PV_TILE
    __syncthreads();   // everyone done reading buffer b before it is re-staged
  }

  // ---- write Out = O / l ----
  float invl[8];
#pragma unroll
  for (int r = 0; r < 8; ++r) invl[r] = 1.0f / lrow[r];
#pragma unroll
  for (int r = 0; r < 8; ++r) {
    const int row = qrow0 + half * 8 + r;
    float* orow = Out + base + (size_t)row * Dn;
    orow[0 * 16 + lm] = o0[r] * invl[r];
    orow[1 * 16 + lm] = o1[r] * invl[r];
    orow[2 * 16 + lm] = o2[r] * invl[r];
    orow[3 * 16 + lm] = o3[r] * invl[r];
  }

  // =========== pass 2: recompute S, stream normalized attn (K only) ===========
  const size_t abase = (size_t)bh * Sn * Sn;
  stage_tile(gK4, &sKb[0][0][0], tid);
  for (int c = 0; c < nchunks; ++c) {
    const int b  = c & 1;
    const int k0 = c * KT;
    if (c + 1 < nchunks) {
      stage_tile(gK4 + (k0 + KT) * (Dn / 4), &sKb[b ^ 1][0][0], tid);
      WAIT_ASYNC(4);
    } else {
      WAIT_ASYNC(0);
    }
    __syncthreads();
#pragma unroll
    for (int t = 0; t < 2; ++t) {
      v16h kb0, kb1;
      const float* kr = &sKb[b][t * 16 + lm][0];
#pragma unroll
      for (int e = 0; e < 16; ++e) {
        kb0[e] = (_Float16)kr[koff + e + (e >= 8 ? 8 : 0)];
        kb1[e] = (_Float16)kr[32 + koff + e + (e >= 8 ? 8 : 0)];
      }
      v8f s = {};
      s = WMMA_F16(qa0, kb0, s);
      s = WMMA_F16(qa1, kb1, s);
#pragma unroll
      for (int r = 0; r < 8; ++r) {
        const int row = qrow0 + half * 8 + r;
        const int kg  = k0 + t * 16 + lm;
        float p = (kg <= row) ? __expf(s[r] - mrow[r]) * invl[r] : 0.0f;
        Attn[abase + (size_t)row * Sn + kg] = p;
      }
    }
    __syncthreads();
  }
}

extern "C" void kernel_launch(void* const* d_in, const int* in_sizes, int n_in,
                              void* d_out, int out_size, void* d_ws, size_t ws_size,
                              hipStream_t stream) {
  (void)in_sizes; (void)n_in; (void)d_ws; (void)ws_size; (void)out_size;
  const float* q = (const float*)d_in[0];
  const float* k = (const float*)d_in[1];
  const float* v = (const float*)d_in[2];
  // d_in[3] is the causal mask; causality is computed analytically in-kernel.

  float* out  = (float*)d_out;
  float* attn = out + (size_t)Bn * Hn * Sn * Dn;

  // zero attn so the masked upper triangle is exactly 0
  const size_t n4 = (size_t)Bn * Hn * Sn * Sn / 4;
  zero_f4_kernel<<<4096, 256, 0, stream>>>((float4*)attn, n4);

  dim3 grid(Sn / QT, Bn * Hn);   // 32 q-tiles x 32 (b,h)
  fa_fwd_kernel<<<grid, 128, 0, stream>>>(q, k, v, out, attn);
}
